// NNCLRLoss_79396765434186
// MI455X (gfx1250) — matrix operand order repaired and checked by reading, not compiled, not run on hardware
//
#include <hip/hip_runtime.h>
#include <hip/hip_bf16.h>

// ---------------------------------------------------------------------------
// NNCLR loss for MI455X (gfx1250, wave32, WMMA 16x16x32 bf16 + async->LDS)
//   inputs : z_i [2048,512] f32, z_j [2048,512] f32, queue [32768,512] f32
//   output : scalar f32
// ---------------------------------------------------------------------------

typedef __bf16 bf16_t;
typedef __attribute__((ext_vector_type(16))) __bf16 v16bf;
typedef __attribute__((ext_vector_type(8)))  __bf16 v8bf;
typedef __attribute__((ext_vector_type(8)))  float  v8f;

#define DDIM 512
#define NF   4096          // 2*B feature rows
#define QN   32768         // queue rows
#define TEMP_INV 2.0f      // 1/0.5
#define NEGINF  (-1e30f)

// block tile 128(M) x 256(N); 8 waves: 2 M-waves x 4 N-waves; wave tile 64x64
#define BM 128
#define BN 256
#define NCH1 (QN / BN)     // 128 column chunks in gemm1
#define NCH3 (NF / BN)     // 16 column chunks in gemm2

// LDS staging for gemm1 (double buffered)
#define KSTEP 32
#define KITERS (DDIM / KSTEP)              // 16
#define A_STRIDE 80                        // 64B data + 16B pad (16B aligned)
#define B_STRIDE 80
#define A_BUF (BM * A_STRIDE)              // 10240 B
#define B_BUF (BN * B_STRIDE)              // 20480 B
#define SMEM_BYTES (2 * (A_BUF + B_BUF))   // 61440 B (< 320KB WGP LDS)
#define A_OFF(s) ((s) * A_BUF)
#define B_OFF(s) (2 * A_BUF + (s) * B_BUF)

#define SHUF16(lo, hi) __builtin_shufflevector(lo, hi, 0,1,2,3,4,5,6,7,8,9,10,11,12,13,14,15)

// --- CDNA5 async global->LDS copy (16B per lane), tracked by ASYNCcnt -------
__device__ __forceinline__ unsigned lds_addr_of(const void* p) {
    // generic -> LDS(as3) addrspacecast; as3 pointers are 32-bit LDS offsets
    return (unsigned)(unsigned long long)
        (__attribute__((address_space(3))) const char*)(const char*)p;
}
__device__ __forceinline__ void async_b128(unsigned lds_byte_off, const void* gsrc) {
    asm volatile("global_load_async_to_lds_b128 %0, %1, off"
                 :: "v"(lds_byte_off), "v"(gsrc) : "memory");
}

// A fragment (16x32 bf16): lane l<16 row=l K=k..k+7 (V0-3), K=k+16..k+23 (V4-7);
// lanes 16-31 same rows, K shifted +8. p points at row start + half*16 bytes.
__device__ __forceinline__ v16bf a_frag_lds(const char* p) {
    v8bf lo = *(const v8bf*)p;
    v8bf hi = *(const v8bf*)(p + 32);
    return SHUF16(lo, hi);
}
// B fragment (32x16 bf16): lane n<16 col=n K=k..k+7, K=k+16..k+23;
// lanes 16-31 K shifted +8. p points at col start + half*16 bytes.
__device__ __forceinline__ v16bf b_frag_lds(const char* p) {
    v8bf lo = *(const v8bf*)p;
    v8bf hi = *(const v8bf*)(p + 32);
    return SHUF16(lo, hi);
}
// direct-from-global variants (row-major source, ld = DDIM elements)
__device__ __forceinline__ v16bf load_a_frag(const bf16_t* p) {
    v8bf lo = *(const v8bf*)p;
    v8bf hi = *(const v8bf*)(p + 16);
    return SHUF16(lo, hi);
}
__device__ __forceinline__ v16bf load_b_frag(const bf16_t* p) {
    v8bf lo = *(const v8bf*)p;
    v8bf hi = *(const v8bf*)(p + 16);
    return SHUF16(lo, hi);
}

// ---------------------------------------------------------------------------
// Kernel 1: L2-normalize rows of [z_i;z_j] and queue, emit bf16 copies.
// ---------------------------------------------------------------------------
__global__ __launch_bounds__(256) void nnclr_normalize(
    const float* __restrict__ zi, const float* __restrict__ zj,
    const float* __restrict__ queue,
    bf16_t* __restrict__ featbf, bf16_t* __restrict__ queuebf) {
    const int row = blockIdx.x;
    const int t   = threadIdx.x;
    const float* src;
    bf16_t* dst;
    if (row < NF) {
        src = (row < NF / 2) ? (zi + (size_t)row * DDIM)
                             : (zj + (size_t)(row - NF / 2) * DDIM);
        dst = featbf + (size_t)row * DDIM;
    } else {
        const int qr = row - NF;
        src = queue + (size_t)qr * DDIM;
        dst = queuebf + (size_t)qr * DDIM;
    }
    const float2 v = ((const float2*)src)[t];
    float ss = v.x * v.x + v.y * v.y;
#pragma unroll
    for (int m = 16; m > 0; m >>= 1) ss += __shfl_xor(ss, m, 32);
    __shared__ float red[8];
    if ((t & 31) == 0) red[t >> 5] = ss;
    __syncthreads();
    float tot = 0.f;
#pragma unroll
    for (int i = 0; i < 8; ++i) tot += red[i];
    const float inv = 1.0f / fmaxf(sqrtf(tot), 1e-12f);
    const bf16_t b0 = (bf16_t)(v.x * inv);
    const bf16_t b1 = (bf16_t)(v.y * inv);
    union { bf16_t h[2]; unsigned int u; } pk;
    pk.h[0] = b0; pk.h[1] = b1;
    ((unsigned int*)dst)[t] = pk.u;
}

// ---------------------------------------------------------------------------
// Kernel 2: sims = feat . queue^T with fused per-row argmax over a 256-col
// chunk. Tiles staged global->LDS via async copies, double buffered.
// grid = (NF/BM, QN/BN) ; 256 threads = 8 waves ; wave tile 64x64.
// ---------------------------------------------------------------------------
__global__ __launch_bounds__(256) void nnclr_gemm1_argmax(
    const bf16_t* __restrict__ featbf, const bf16_t* __restrict__ queuebf,
    float* __restrict__ pmax, int* __restrict__ pidx) {
    extern __shared__ char smem[];
    const int tid  = threadIdx.x;
    const int wave = tid >> 5;
    const int lane = tid & 31;
    const int half = lane >> 4;
    const int l    = lane & 15;
    const int rowblk = blockIdx.x * BM;
    const int colblk = blockIdx.y * BN;
    const int qch    = blockIdx.y;
    const int mwave  = (wave & 1) * 64;   // wave row offset in block tile
    const int nwave  = (wave >> 1) * 64;  // wave col offset in block tile

    // per-thread async-copy slots: A has 512 16B-chunks (2/thread),
    // B has 1024 (4/thread). chunk c -> (row = c>>2, sub = c&3).
    unsigned alds[2]; const bf16_t* agsrc[2];
    unsigned blds[4]; const bf16_t* bgsrc[4];
#pragma unroll
    for (int i = 0; i < 2; ++i) {
        const int c = 2 * tid + i, r = c >> 2, sub = c & 3;
        alds[i]  = (unsigned)(r * A_STRIDE + sub * 16);
        agsrc[i] = featbf + (size_t)(rowblk + r) * DDIM + sub * 8;
    }
#pragma unroll
    for (int i = 0; i < 4; ++i) {
        const int c = 4 * tid + i, r = c >> 2, sub = c & 3;
        blds[i]  = (unsigned)(r * B_STRIDE + sub * 16);
        bgsrc[i] = queuebf + (size_t)(colblk + r) * DDIM + sub * 8;
    }
    const unsigned sbase = lds_addr_of(smem);

    // prologue: issue stage 0 (6 async ops per wave)
#pragma unroll
    for (int i = 0; i < 2; ++i) async_b128(sbase + A_OFF(0) + alds[i], agsrc[i]);
#pragma unroll
    for (int i = 0; i < 4; ++i) async_b128(sbase + B_OFF(0) + blds[i], bgsrc[i]);

    const v8f zero = {0.f, 0.f, 0.f, 0.f, 0.f, 0.f, 0.f, 0.f};
    v8f acc[4][4];
#pragma unroll
    for (int mt = 0; mt < 4; ++mt)
#pragma unroll
        for (int nt = 0; nt < 4; ++nt) acc[mt][nt] = zero;

    for (int ks = 0; ks < KITERS; ++ks) {
        const int cur = ks & 1;
        if (ks + 1 < KITERS) {
            const int nxt  = cur ^ 1;
            const int koff = (ks + 1) * KSTEP;   // element offset into K
#pragma unroll
            for (int i = 0; i < 2; ++i)
                async_b128(sbase + A_OFF(nxt) + alds[i], agsrc[i] + koff);
#pragma unroll
            for (int i = 0; i < 4; ++i)
                async_b128(sbase + B_OFF(nxt) + blds[i], bgsrc[i] + koff);
            // 6 outstanding from next stage allowed; stage ks complete (in-order)
            asm volatile("s_wait_asynccnt 0x6" ::: "memory");
        } else {
            asm volatile("s_wait_asynccnt 0x0" ::: "memory");
        }
        __syncthreads();   // all waves' stage-ks tiles visible in LDS

        const char* abase = smem + A_OFF(cur) + half * 16;
        const char* bbase = smem + B_OFF(cur) + half * 16;
        v16bf a[4];
#pragma unroll
        for (int mt = 0; mt < 4; ++mt)
            a[mt] = a_frag_lds(abase + (mwave + mt * 16 + l) * A_STRIDE);
#pragma unroll
        for (int nt = 0; nt < 4; ++nt) {
            const v16bf b = b_frag_lds(bbase + (nwave + nt * 16 + l) * B_STRIDE);
#pragma unroll
            for (int mt = 0; mt < 4; ++mt)
                acc[mt][nt] = __builtin_amdgcn_wmma_f32_16x16x32_bf16(
                    false, a[mt], false, b, (short)0, acc[mt][nt], false, false);
        }
        __syncthreads();   // done reading buf[cur] before it is overwritten
    }

    // per-row argmax: VGPR j holds row (mt*16 + j + 8*half); 16 lanes of a
    // half-group hold 16 consecutive columns. Branchless merges.
    const int rowbase = rowblk + mwave;
    const int colbase = colblk + nwave;
#pragma unroll
    for (int mt = 0; mt < 4; ++mt) {
#pragma unroll
        for (int j = 0; j < 8; ++j) {
            float best = acc[mt][0][j];
            int   bcol = colbase + l;
#pragma unroll
            for (int nt = 1; nt < 4; ++nt) {
                const float v = acc[mt][nt][j];
                const int   c = colbase + nt * 16 + l;
                const bool  g = (v > best) || (v == best && c < bcol);
                best = g ? v : best;
                bcol = g ? c : bcol;
            }
#pragma unroll
            for (int m = 1; m < 16; m <<= 1) {
                const float ov = __shfl_xor(best, m, 32);
                const int   oc = __shfl_xor(bcol, m, 32);
                const bool  g  = (ov > best) || (ov == best && oc < bcol);
                best = g ? ov : best;
                bcol = g ? oc : bcol;
            }
            if (l == j) {
                const int row = rowbase + mt * 16 + j + 8 * half;
                pmax[(size_t)row * NCH1 + qch] = best;
                pidx[(size_t)row * NCH1 + qch] = bcol;
            }
        }
    }
}

// ---------------------------------------------------------------------------
// Kernel 3: final argmax over chunk partials + gather queue row -> nn_feats.
// ---------------------------------------------------------------------------
__global__ __launch_bounds__(256) void nnclr_argmax_gather(
    const float* __restrict__ pmax, const int* __restrict__ pidx,
    const bf16_t* __restrict__ queuebf, bf16_t* __restrict__ nnbf) {
    const int row = blockIdx.x;
    const int t   = threadIdx.x;
    float v = NEGINF;
    int   c = 0x7fffffff;
    if (t < NCH1) { v = pmax[(size_t)row * NCH1 + t]; c = pidx[(size_t)row * NCH1 + t]; }
#pragma unroll
    for (int m = 1; m < 32; m <<= 1) {
        const float ov = __shfl_xor(v, m, 32);
        const int   oc = __shfl_xor(c, m, 32);
        const bool  g  = (ov > v) || (ov == v && oc < c);
        v = g ? ov : v;
        c = g ? oc : c;
    }
    __shared__ float wv[8];
    __shared__ int   wc[8];
    if ((t & 31) == 0) { wv[t >> 5] = v; wc[t >> 5] = c; }
    __syncthreads();
    if (t == 0) {
        float bv = wv[0]; int bc = wc[0];
#pragma unroll
        for (int i = 1; i < 8; ++i) {
            const bool g = (wv[i] > bv) || (wv[i] == bv && wc[i] < bc);
            bv = g ? wv[i] : bv;
            bc = g ? wc[i] : bc;
        }
        wc[0] = bc;
    }
    __syncthreads();
    const int best = wc[0];
    ((unsigned int*)(nnbf + (size_t)row * DDIM))[t] =
        ((const unsigned int*)(queuebf + (size_t)best * DDIM))[t];
}

// ---------------------------------------------------------------------------
// Kernel 4: sim = nn . feat^T / T ; fused diag mask, positive capture,
// per-row streaming logsumexp partials over a 256-column chunk.
// ---------------------------------------------------------------------------
__global__ __launch_bounds__(256) void nnclr_gemm2_lse(
    const bf16_t* __restrict__ nnbf, const bf16_t* __restrict__ featbf,
    float* __restrict__ pm2, float* __restrict__ ps2, float* __restrict__ pos) {
    const int wave = threadIdx.x >> 5;
    const int lane = threadIdx.x & 31;
    const int half = lane >> 4;
    const int l    = lane & 15;
    const int rowbase = blockIdx.x * BM + (wave & 1) * 64;
    const int colbase = blockIdx.y * BN + (wave >> 1) * 64;
    const int nch = blockIdx.y;

    const v8f zero = {0.f, 0.f, 0.f, 0.f, 0.f, 0.f, 0.f, 0.f};
    v8f acc[4][4];
#pragma unroll
    for (int mt = 0; mt < 4; ++mt)
#pragma unroll
        for (int nt = 0; nt < 4; ++nt) acc[mt][nt] = zero;

    const bf16_t* aptr = nnbf   + (size_t)(rowbase + l) * DDIM + half * 8;
    const bf16_t* bptr = featbf + (size_t)(colbase + l) * DDIM + half * 8;

    for (int k0 = 0; k0 < DDIM; k0 += 32) {
        v16bf a[4];
#pragma unroll
        for (int mt = 0; mt < 4; ++mt)
            a[mt] = load_a_frag(aptr + (size_t)(mt * 16) * DDIM + k0);
#pragma unroll
        for (int nt = 0; nt < 4; ++nt) {
            const v16bf b = load_b_frag(bptr + (size_t)(nt * 16) * DDIM + k0);
#pragma unroll
            for (int mt = 0; mt < 4; ++mt)
                acc[mt][nt] = __builtin_amdgcn_wmma_f32_16x16x32_bf16(
                    false, a[mt], false, b, (short)0, acc[mt][nt], false, false);
        }
    }

#pragma unroll
    for (int mt = 0; mt < 4; ++mt) {
#pragma unroll
        for (int j = 0; j < 8; ++j) {
            const int row  = rowbase + mt * 16 + j + 8 * half;
            const int pcol = row ^ (NF / 2);   // (row + N/2) mod N
            float v[4];
#pragma unroll
            for (int nt = 0; nt < 4; ++nt) {
                float x = TEMP_INV * acc[mt][nt][j];
                const int c = colbase + nt * 16 + l;
                if (c == pcol) pos[row] = x;        // unique (row,col) writer
                v[nt] = (c == row) ? NEGINF : x;    // mask diagonal
            }
            float m = fmaxf(fmaxf(v[0], v[1]), fmaxf(v[2], v[3]));
            float s = __expf(v[0] - m) + __expf(v[1] - m) +
                      __expf(v[2] - m) + __expf(v[3] - m);
#pragma unroll
            for (int msk = 1; msk < 16; msk <<= 1) {
                const float om = __shfl_xor(m, msk, 32);
                const float os = __shfl_xor(s, msk, 32);
                const float M  = fmaxf(m, om);
                s = s * __expf(m - M) + os * __expf(om - M);
                m = M;
            }
            if (l == j) {
                pm2[(size_t)row * NCH3 + nch] = m;
                ps2[(size_t)row * NCH3 + nch] = s;
            }
        }
    }
}

// ---------------------------------------------------------------------------
// Kernel 5: merge lse partials per row, reduce means, write the scalar loss.
// ---------------------------------------------------------------------------
__global__ __launch_bounds__(256) void nnclr_finalize(
    const float* __restrict__ pm2, const float* __restrict__ ps2,
    const float* __restrict__ pos, float* __restrict__ out) {
    const int t = threadIdx.x;
    float sum_lse = 0.f, sum_pos = 0.f;
    for (int r = t; r < NF; r += 256) {
        float M = pm2[(size_t)r * NCH3];
        float S = ps2[(size_t)r * NCH3];
#pragma unroll
        for (int i = 1; i < NCH3; ++i) {
            const float m2 = pm2[(size_t)r * NCH3 + i];
            const float s2 = ps2[(size_t)r * NCH3 + i];
            const float Mn = fmaxf(M, m2);
            S = S * __expf(M - Mn) + s2 * __expf(m2 - Mn);
            M = Mn;
        }
        sum_lse += M + logf(S);
        sum_pos += pos[r];
    }
#pragma unroll
    for (int m = 16; m > 0; m >>= 1) {
        sum_lse += __shfl_xor(sum_lse, m, 32);
        sum_pos += __shfl_xor(sum_pos, m, 32);
    }
    __shared__ float rl[8], rp[8];
    if ((t & 31) == 0) { rl[t >> 5] = sum_lse; rp[t >> 5] = sum_pos; }
    __syncthreads();
    if (t == 0) {
        float L = 0.f, P = 0.f;
#pragma unroll
        for (int i = 0; i < 8; ++i) { L += rl[i]; P += rp[i]; }
        out[0] = (L - P) / (float)NF;   // mean(lse) - mean(pos)
    }
}

// ---------------------------------------------------------------------------
extern "C" void kernel_launch(void* const* d_in, const int* in_sizes, int n_in,
                              void* d_out, int out_size, void* d_ws, size_t ws_size,
                              hipStream_t stream) {
    (void)in_sizes; (void)n_in; (void)out_size; (void)ws_size;
    const float* zi    = (const float*)d_in[0];
    const float* zj    = (const float*)d_in[1];
    const float* queue = (const float*)d_in[2];
    float* out = (float*)d_out;

    char* w = (char*)d_ws;
    bf16_t* featbf  = (bf16_t*)w; w += (size_t)NF * DDIM * 2;   //  4 MB
    bf16_t* queuebf = (bf16_t*)w; w += (size_t)QN * DDIM * 2;   // 32 MB
    bf16_t* nnbf    = (bf16_t*)w; w += (size_t)NF * DDIM * 2;   //  4 MB
    float*  pmax    = (float*)w;  w += (size_t)NF * NCH1 * 4;   //  2 MB
    int*    pidx    = (int*)w;    w += (size_t)NF * NCH1 * 4;   //  2 MB
    float*  pm2     = (float*)w;  w += (size_t)NF * NCH3 * 4;   // 256 KB
    float*  ps2     = (float*)w;  w += (size_t)NF * NCH3 * 4;   // 256 KB
    float*  pos     = (float*)w;  w += (size_t)NF * 4;          //  16 KB

    nnclr_normalize<<<NF + QN, 256, 0, stream>>>(zi, zj, queue, featbf, queuebf);

    dim3 g1(NF / BM, QN / BN);   // 32 x 128
    nnclr_gemm1_argmax<<<g1, 256, SMEM_BYTES, stream>>>(featbf, queuebf, pmax, pidx);

    nnclr_argmax_gather<<<NF, 256, 0, stream>>>(pmax, pidx, queuebf, nnbf);

    dim3 g2(NF / BM, NF / BN);   // 32 x 16
    nnclr_gemm2_lse<<<g2, 256, 0, stream>>>(nnbf, featbf, pm2, ps2, pos);

    nnclr_finalize<<<1, 256, 0, stream>>>(pm2, ps2, pos, out);
}
